// Decoder_x_82411832475896
// MI455X (gfx1250) — matrix operand order repaired and checked by reading, not compiled, not run on hardware
//
#include <hip/hip_runtime.h>

// Problem constants (match reference)
#define B_   1024
#define ZD_  256
#define HD_  1024
#define L_   32
#define NW_  64
#define XD_  1024
#define H3_  (3 * HD_)
#define KX_  (ZD_ + 1)   // 257 (row stride of Wih)

typedef __attribute__((ext_vector_type(16))) __bf16         v16bf;
typedef __attribute__((ext_vector_type(8)))  unsigned int   v8u;
typedef __attribute__((ext_vector_type(8)))  float          v8f;

// Pack two f32 -> bf16x2 (low = first arg), round-to-nearest-even, then one
// v_perm_b32 splices the two high halves (no half-register mov chains).
__device__ __forceinline__ unsigned int pack_bf16x2(float lo, float hi) {
    unsigned int ulo = __float_as_uint(lo);
    unsigned int uhi = __float_as_uint(hi);
    ulo += 0x7FFFu + ((ulo >> 16) & 1u);
    uhi += 0x7FFFu + ((uhi >> 16) & 1u);
    // result bytes: [1:0] = ulo bytes [3:2], [3:2] = uhi bytes [3:2]
    return __builtin_amdgcn_perm(uhi, ulo, 0x07060302u);
}

__device__ __forceinline__ float sigmoidf_(float x) {
    return 1.0f / (1.0f + expf(-x));
}

// ---------------------------------------------------------------------------
// C = act(A[M,K] @ W[N,K]^T + bias[N]), arbitrary output strides, NO guards:
// M multiple of 128, N multiple of 64, K multiple of 32 (launcher guarantees).
// Block: 128 threads = 4 wave32s (2x2). Block tile 128x64, wave tile 64x32:
// 4 A-frags x 2 B-frags -> 8 x v_wmma_f32_16x16x32_bf16 per K-step.
// Double-buffered LDS, one barrier per K-step.
// WVEC: W rows 16B-aligned (ldw % 4 == 0) -> float4 loads, else scalar.
// ACT: 0=none, 1=relu, 2=sigmoid
// ---------------------------------------------------------------------------
#define BM 128
#define BN 64
#define BK 32

template <bool WVEC, int ACT>
__global__ __launch_bounds__(128)
void gemm_bf16_wmma(const float* __restrict__ A, int lda,
                    const float* __restrict__ W, int ldw,
                    const float* __restrict__ bias,
                    float* __restrict__ C, long long crs, long long ccs,
                    int K)
{
    // bf16 pairs: [row][pair]; row = 32 K-values = 16 uints = 64B
    __shared__ unsigned int As[2][BM][BK / 2];   // 2 x 8 KB
    __shared__ unsigned int Bs[2][BN][BK / 2];   // 2 x 4 KB

    const int tid  = threadIdx.x;
    const int lane = tid & 31;
    const int wave = tid >> 5;           // 0..3
    const int wm   = (wave >> 1) * 64;   // wave M offset within block tile
    const int wn   = (wave & 1)  * 32;   // wave N offset within block tile

    const int m0 = blockIdx.y * BM;
    const int n0 = blockIdx.x * BN;

    // Loader mapping. A: 1 thread per row (32 floats). W: 2 threads per row.
    const int wrow = tid >> 1;           // 0..63
    const int wlk  = (tid & 1) * 16;     // 0 or 16

    v8f acc[4][2];
    #pragma unroll
    for (int i = 0; i < 4; i++)
        #pragma unroll
        for (int j = 0; j < 2; j++)
            #pragma unroll
            for (int e = 0; e < 8; e++)
                acc[i][j][e] = 0.0f;

    auto load_tile = [&](int k0, int bs) {
        // ---- A tile: 128 rows x 32 K, one row per thread ----
        {
            const float* src = A + (size_t)(m0 + tid) * lda + k0;
            float4 a[8];
            #pragma unroll
            for (int i = 0; i < 8; i++) a[i] = *(const float4*)(src + 4 * i);
            unsigned int* dst = &As[bs][tid][0];
            #pragma unroll
            for (int i = 0; i < 4; i++) {
                uint4 p = { pack_bf16x2(a[2 * i].x,     a[2 * i].y),
                            pack_bf16x2(a[2 * i].z,     a[2 * i].w),
                            pack_bf16x2(a[2 * i + 1].x, a[2 * i + 1].y),
                            pack_bf16x2(a[2 * i + 1].z, a[2 * i + 1].w) };
                *(uint4*)(dst + 4 * i) = p;
            }
        }
        // ---- W tile: 64 rows x 32 K, two threads per row ----
        {
            const float* src = W + (size_t)(n0 + wrow) * ldw + (k0 + wlk);
            float w[16];
            if constexpr (WVEC) {
                *(float4*)&w[0]  = *(const float4*)(src + 0);
                *(float4*)&w[4]  = *(const float4*)(src + 4);
                *(float4*)&w[8]  = *(const float4*)(src + 8);
                *(float4*)&w[12] = *(const float4*)(src + 12);
            } else {
                #pragma unroll
                for (int i = 0; i < 16; i++) w[i] = src[i];
            }
            uint4 p0 = { pack_bf16x2(w[0],  w[1]),  pack_bf16x2(w[2],  w[3]),
                         pack_bf16x2(w[4],  w[5]),  pack_bf16x2(w[6],  w[7]) };
            uint4 p1 = { pack_bf16x2(w[8],  w[9]),  pack_bf16x2(w[10], w[11]),
                         pack_bf16x2(w[12], w[13]), pack_bf16x2(w[14], w[15]) };
            unsigned int* dst = &Bs[bs][wrow][wlk >> 1];
            *(uint4*)(dst + 0) = p0;
            *(uint4*)(dst + 4) = p1;
        }
    };

    const int kh = lane >> 4;    // K half selector per ISA fragment layout
    const int lr = lane & 15;

    const int nt = K >> 5;       // K multiple of 32
    load_tile(0, 0);

    for (int i = 0; i < nt; i++) {
        __syncthreads();
        if (i + 1 < nt) load_tile((i + 1) << 5, (i + 1) & 1);
        const int cur = i & 1;

        // B fragments first (shared across all fm): elements 0..7 = K[8h..8h+7], 8..15 = K[16+8h..16+8h+7]
        v8u bfr[2];
        #pragma unroll
        for (int fn = 0; fn < 2; fn++) {
            const unsigned int* rowp = &Bs[cur][wn + fn * 16 + lr][0];
            uint4 c0 = *(const uint4*)(rowp + 4 * kh);
            uint4 c1 = *(const uint4*)(rowp + 8 + 4 * kh);
            bfr[fn] = (v8u){ c0.x, c0.y, c0.z, c0.w, c1.x, c1.y, c1.z, c1.w };
        }
        // A fragments: elements 0..7 = K[8h..8h+7], 8..15 = K[16+8h..16+8h+7]
        #pragma unroll
        for (int fm = 0; fm < 4; fm++) {
            const unsigned int* rowp = &As[cur][wm + fm * 16 + lr][0];
            uint4 c0 = *(const uint4*)(rowp + 4 * kh);
            uint4 c1 = *(const uint4*)(rowp + 8 + 4 * kh);
            v8u afr = (v8u){ c0.x, c0.y, c0.z, c0.w, c1.x, c1.y, c1.z, c1.w };
            #pragma unroll
            for (int fn = 0; fn < 2; fn++)
                acc[fm][fn] = __builtin_amdgcn_wmma_f32_16x16x32_bf16(
                    false, __builtin_bit_cast(v16bf, afr),
                    false, __builtin_bit_cast(v16bf, bfr[fn]),
                    (short)0, acc[fm][fn], false, false);
        }
    }

    // ---- epilogue: bias + activation + strided store (no guards) ----
    #pragma unroll
    for (int fm = 0; fm < 4; fm++) {
        #pragma unroll
        for (int fn = 0; fn < 2; fn++) {
            const int n = n0 + wn + fn * 16 + lr;
            const float bn = bias ? bias[n] : 0.0f;
            #pragma unroll
            for (int v = 0; v < 8; v++) {
                const int m = m0 + wm + fm * 16 + v + 8 * kh;
                float val = acc[fm][fn][v] + bn;
                if constexpr (ACT == 1) val = val > 0.0f ? val : 0.0f;
                if constexpr (ACT == 2) val = sigmoidf_(val);
                C[(long long)m * crs + (long long)n * ccs] = val;
            }
        }
    }
}

// ---------------------------------------------------------------------------
// Elementwise kernels
// ---------------------------------------------------------------------------
__global__ void zero_h_kernel(float4* __restrict__ h4) {
    int idx = blockIdx.x * blockDim.x + threadIdx.x;
    if (idx < B_ * HD_ / 4) h4[idx] = float4{0.f, 0.f, 0.f, 0.f};
}

// GRU combine, 4 elements per thread. Adds the folded rank-1 term
// t_l[b] * Wih[:,256] and bih (GX was computed without them), then
// r/u/n gates; h <- z_theta in place (z_theta is the scan carry).
__global__ void gru_combine_kernel(const float* __restrict__ GX,
                                   const float* __restrict__ GH,
                                   float* __restrict__ h,
                                   const float* __restrict__ eps_l,
                                   const float* __restrict__ t, int l,
                                   const float* __restrict__ WihL,   // [3HD, 257]
                                   const float* __restrict__ bihL)   // [3HD]
{
    int idx = blockIdx.x * blockDim.x + threadIdx.x;
    if (idx >= B_ * HD_ / 4) return;
    const int b = idx / (HD_ / 4);
    const int j = (idx % (HD_ / 4)) * 4;
    const float tb = t[b * L_ + l];

    const float* gx = GX + (size_t)b * H3_;
    const float* gh = GH + (size_t)b * H3_;
    float4 xr = *(const float4*)(gx + j);
    float4 xu = *(const float4*)(gx + HD_ + j);
    float4 xn = *(const float4*)(gx + 2 * HD_ + j);
    float4 hr = *(const float4*)(gh + j);
    float4 hu = *(const float4*)(gh + HD_ + j);
    float4 hn = *(const float4*)(gh + 2 * HD_ + j);
    float4 hv = *(const float4*)(h + (size_t)b * HD_ + j);
    float4 ep = *(const float4*)(eps_l + (size_t)b * HD_ + j);

    const float xrr[4] = {xr.x, xr.y, xr.z, xr.w};
    const float xuu[4] = {xu.x, xu.y, xu.z, xu.w};
    const float xnv[4] = {xn.x, xn.y, xn.z, xn.w};
    const float hrr[4] = {hr.x, hr.y, hr.z, hr.w};
    const float huu[4] = {hu.x, hu.y, hu.z, hu.w};
    const float hnn[4] = {hn.x, hn.y, hn.z, hn.w};
    const float hvv[4] = {hv.x, hv.y, hv.z, hv.w};
    const float epp[4] = {ep.x, ep.y, ep.z, ep.w};

    const float kStd = 1.0025031276057952f;   // exp(0.5 * 0.005)
    float out[4];
    #pragma unroll
    for (int i = 0; i < 4; i++) {
        const int jj = j + i;
        const float wr = WihL[(size_t)(jj)           * KX_ + ZD_];
        const float wu = WihL[(size_t)(HD_ + jj)     * KX_ + ZD_];
        const float wn = WihL[(size_t)(2 * HD_ + jj) * KX_ + ZD_];
        const float fxr = xrr[i] + tb * wr + bihL[jj];
        const float fxu = xuu[i] + tb * wu + bihL[HD_ + jj];
        const float fxn = xnv[i] + tb * wn + bihL[2 * HD_ + jj];
        const float r = sigmoidf_(fxr + hrr[i]);
        const float u = sigmoidf_(fxu + huu[i]);
        const float nn = tanhf(fxn + r * hnn[i]);
        const float mu = (1.0f - u) * nn + u * hvv[i];
        out[i] = mu + kStd * epp[i];
    }
    *(float4*)(h + (size_t)b * HD_ + j) = float4{out[0], out[1], out[2], out[3]};
}

// ---------------------------------------------------------------------------
// Orchestration
// ---------------------------------------------------------------------------
extern "C" void kernel_launch(void* const* d_in, const int* in_sizes, int n_in,
                              void* d_out, int out_size, void* d_ws, size_t ws_size,
                              hipStream_t stream)
{
    const float* z    = (const float*)d_in[0];
    const float* t    = (const float*)d_in[1];
    const float* eps  = (const float*)d_in[2];
    const float* Wih  = (const float*)d_in[3];
    const float* Whh  = (const float*)d_in[4];
    const float* bih  = (const float*)d_in[5];
    const float* bhh  = (const float*)d_in[6];
    const float* pW1  = (const float*)d_in[7];
    const float* pb1  = (const float*)d_in[8];
    const float* pW2  = (const float*)d_in[9];
    const float* pb2  = (const float*)d_in[10];
    const float* recW = (const float*)d_in[11];
    const float* recb = (const float*)d_in[12];

    float* out = (float*)d_out;

    // Workspace layout (floats)
    float* ws = (float*)d_ws;
    float* GX = ws;                                   // [B, 3HD]
    float* GH = GX + (size_t)B_ * H3_;                // [B, 3HD]
    float* h  = GH + (size_t)B_ * H3_;                // [B, HD]
    float* A1 = h  + (size_t)B_ * HD_;                // [B, HD]

    const dim3 blk1(256);
    zero_h_kernel<<<(B_ * HD_ / 4 + 255) / 256, blk1, 0, stream>>>((float4*)h);

    const dim3 gblk(128);
    const dim3 g_gates(H3_ / BN, B_ / BM);   // 48 x 8
    const dim3 g_hd   (HD_ / BN, B_ / BM);   // 16 x 8
    const dim3 g_nw   (NW_ / BN, B_ / BM);   //  1 x 8
    const int  n_ew   = (B_ * HD_ / 4 + 255) / 256;

    for (int l = 0; l < L_; l++) {
        const float* WihL = Wih + (size_t)l * H3_ * KX_;

        // GX = z @ Wih_l[:, :256]^T   (K=256, no bias; t/bih folded later)
        gemm_bf16_wmma<false, 0><<<g_gates, gblk, 0, stream>>>(
            z, ZD_, WihL, KX_, nullptr,
            GX, (long long)H3_, 1LL, ZD_);

        // GH = h @ Whh_l^T + bhh_l
        gemm_bf16_wmma<true, 0><<<g_gates, gblk, 0, stream>>>(
            h, HD_, Whh + (size_t)l * H3_ * HD_, HD_, bhh + (size_t)l * H3_,
            GH, (long long)H3_, 1LL, HD_);

        // GRU elementwise: h <- z_theta (in place)
        gru_combine_kernel<<<n_ew, blk1, 0, stream>>>(
            GX, GH, h, eps + (size_t)l * B_ * HD_, t, l,
            WihL, bih + (size_t)l * H3_);

        // A1 = relu(h @ pW1_l^T + pb1_l)
        gemm_bf16_wmma<true, 1><<<g_hd, gblk, 0, stream>>>(
            h, HD_, pW1 + (size_t)l * HD_ * HD_, HD_, pb1 + (size_t)l * HD_,
            A1, (long long)HD_, 1LL, HD_);

        // pred = sigmoid(A1 @ pW2_l^T + pb2_l) scattered into
        // workers_preds[w, b, l] at out[B*XD + w*B*L + b*L + l]
        gemm_bf16_wmma<true, 2><<<g_nw, gblk, 0, stream>>>(
            A1, HD_, pW2 + (size_t)l * NW_ * HD_, HD_, pb2 + (size_t)l * NW_,
            out + (size_t)B_ * XD_ + l, (long long)L_, (long long)(B_ * L_),
            HD_);
    }

    // recon = sigmoid(hT @ recW^T + recb)  -> out[0 : B*XD]
    gemm_bf16_wmma<true, 2><<<g_hd, gblk, 0, stream>>>(
        h, HD_, recW, HD_, recb,
        out, (long long)XD_, 1LL, HD_);
}
